// DinoV2ViT_16982300688695
// MI455X (gfx1250) — compile-verified, matrix operand order checked by reading, not run on hardware
//
#include <hip/hip_runtime.h>
#include <hip/hip_bf16.h>
#include <math.h>

#define DIMC   384
#define HEADS  6
#define HD     64
#define DEPTH  12
#define PATCHK 14
#define IMG    518
#define GRIDP  37
#define NSP    1369
#define NTOK   1370
#define NPAD   1376       // NTOK padded to multiple of 32
#define HID    1536
#define BATCH  2
#define MROWS  (BATCH*NTOK)   // 2740
#define LNEPS  1e-5f
#define ATTN_LDS (16*NPAD*4 + 16*NPAD*2)  // fp32 scores + bf16 P = 132096 B

typedef __bf16 bf16;
typedef __attribute__((ext_vector_type(16))) __bf16 v16bf;
typedef __attribute__((ext_vector_type(8)))  float  v8f;

union FragU { v16bf v; uint4 q[2]; };
union BfBits { unsigned short u; bf16 b; };

__device__ inline bf16 f2bf(float f) {
  unsigned u = __float_as_uint(f);
  unsigned r = (u + 0x7FFFu + ((u >> 16) & 1u)) >> 16;   // RNE
  BfBits t; t.u = (unsigned short)r;
  return t.b;
}

// CDNA5 async global->LDS staging (ASYNCcnt-tracked); guarded fallback.
#if defined(__gfx1250__) && __has_builtin(__builtin_amdgcn_global_load_async_to_lds_b128) && \
    __has_builtin(__builtin_amdgcn_s_wait_asynccnt)
#define USE_ASYNC 1
typedef int v4i_ __attribute__((vector_size(16)));
typedef __attribute__((address_space(1))) v4i_ as1_v4i;   // global
typedef __attribute__((address_space(3))) v4i_ as3_v4i;   // LDS
#else
#define USE_ASYNC 0
#endif

// ---------------------------------------------------------------- fp32 -> bf16
__global__ void cvt_kernel(const float* __restrict__ src, bf16* __restrict__ dst, long n) {
  long i = (long)blockIdx.x * blockDim.x + threadIdx.x;
  long stride = (long)gridDim.x * blockDim.x;
  for (; i < n; i += stride) dst[i] = f2bf(src[i]);
}

__global__ void copy_kernel(const float* __restrict__ src, float* __restrict__ dst, long n) {
  long i = (long)blockIdx.x * blockDim.x + threadIdx.x;
  long stride = (long)gridDim.x * blockDim.x;
  for (; i < n; i += stride) dst[i] = src[i];
}

// ---------------------------------------------------------------- patch embed
__global__ void patch_kernel(const float* __restrict__ x, const float* __restrict__ pw,
                             const float* __restrict__ pbias, const float* __restrict__ pos,
                             float* __restrict__ h) {
  int p = blockIdx.x, cg = blockIdx.y, b = blockIdx.z;
  int py = p / GRIDP, px = p % GRIDP;
  __shared__ float xs[3 * PATCHK * PATCHK];
  for (int idx = threadIdx.x; idx < 3 * PATCHK * PATCHK; idx += 128) {
    int ci = idx / (PATCHK * PATCHK);
    int rem = idx - ci * PATCHK * PATCHK;
    int kh = rem / PATCHK, kw = rem - kh * PATCHK;
    xs[idx] = x[((long)(b * 3 + ci) * IMG + py * PATCHK + kh) * IMG + px * PATCHK + kw];
  }
  __syncthreads();
  int c = cg * 128 + threadIdx.x;
  const float* w = pw + (long)c * (3 * PATCHK * PATCHK);
  float acc = pbias[c];
  for (int t = 0; t < 3 * PATCHK * PATCHK; ++t) acc = fmaf(xs[t], w[t], acc);
  acc += pos[(long)(1 + p) * DIMC + c];
  h[((long)(b * NTOK) + 1 + p) * DIMC + c] = acc;
}

__global__ void cls_kernel(const float* __restrict__ cls, const float* __restrict__ pos,
                           float* __restrict__ h) {
  int b = blockIdx.x, c = threadIdx.x;
  h[(long)(b * NTOK) * DIMC + c] = cls[c] + pos[c];
}

// ---------------------------------------------------------------- LayerNorm
__global__ void ln_kernel(float* __restrict__ h, const float* __restrict__ g,
                          const float* __restrict__ bta, bf16* __restrict__ out,
                          const float* __restrict__ cam, int set_cam) {
  int rowid = blockIdx.x;
  int tid = threadIdx.x;
  bool iscam = set_cam && ((rowid % NTOK) == 0);
  long base = (long)rowid * DIMC;
  float vals[3], s = 0.f, sq = 0.f;
  for (int i = 0; i < 3; ++i) {
    int c = tid + i * 128;
    float v;
    if (iscam) { v = cam[c]; h[base + c] = v; } else v = h[base + c];
    vals[i] = v; s += v; sq += v * v;
  }
  for (int o = 16; o >= 1; o >>= 1) { s += __shfl_xor(s, o); sq += __shfl_xor(sq, o); }
  __shared__ float rs[4], rq[4];
  int wv = tid >> 5, ln = tid & 31;
  if (ln == 0) { rs[wv] = s; rq[wv] = sq; }
  __syncthreads();
  s = rs[0] + rs[1] + rs[2] + rs[3];
  sq = rq[0] + rq[1] + rq[2] + rq[3];
  float mean = s * (1.f / DIMC);
  float var = sq * (1.f / DIMC) - mean * mean;
  float rstd = rsqrtf(var + LNEPS);
  for (int i = 0; i < 3; ++i) {
    int c = tid + i * 128;
    out[base + c] = f2bf((vals[i] - mean) * rstd * g[c] + bta[c]);
  }
}

// ---------------------------------------------------------------- generic GEMM
// C[M,Nout] = A[M,K](bf16) x W[Nout,K](bf16)^T via v_wmma_f32_16x16x32_bf16.
// A tile (64 x 64-K-chunk) is double-buffer staged in LDS via async loads.
// mode 0: outf = acc+bias   mode 1: outb = bf16(gelu)   mode 2: resid += v*ls
#define KC 64
#define ASTRIDE 144                 // bytes per staged A row (72 bf16, padded)
#define ABUF (64 * ASTRIDE)         // 9216 B per buffer

__global__ void gemm_kernel(const bf16* __restrict__ A, const bf16* __restrict__ W,
                            int M, int K, int Nout,
                            const float* __restrict__ bias, const float* __restrict__ ls,
                            float* __restrict__ outf, bf16* __restrict__ outb,
                            float* __restrict__ resid, int mode) {
  __shared__ char asmem[2 * ABUF];
  int tid = threadIdx.x;
  int wave = tid >> 5, lane = tid & 31;
  int wm = wave >> 1, wn = wave & 1;            // 8 waves: 4 M-tiles x 2 N-slabs
  int rb = blockIdx.x * 64;
  int r0 = rb + wm * 16;
  int c0 = blockIdx.y * 128 + wn * 64;
  int col = lane & 15, khalf = lane >> 4;
  int srow = tid >> 3;                          // staging: row 0..31 (+32)
  int soff = (tid & 7) * 8;                     // staging: bf16 offset in row

  auto stage = [&](int buf, int kc) {
#pragma unroll
    for (int rd = 0; rd < 2; ++rd) {
      int r = srow + rd * 32;
      int grow = rb + r; if (grow > M - 1) grow = M - 1;     // clamp edge rows
      const bf16* gp = A + (long)grow * K + kc + soff;
      char* lp = asmem + buf * ABUF + r * ASTRIDE + soff * 2;
#if USE_ASYNC
      __builtin_amdgcn_global_load_async_to_lds_b128((as1_v4i*)gp, (as3_v4i*)lp, 0, 0);
#else
      *(uint4*)lp = *(const uint4*)gp;
#endif
    }
  };

  int nkc = K / KC;                             // K in {384,1536}: divisible
  stage(0, 0);
  v8f acc[4] = {};
  for (int i = 0; i < nkc; ++i) {
    int kc = i * KC;
    if (i + 1 < nkc) {
      stage((i + 1) & 1, kc + KC);
      __builtin_prefetch(W + (long)(c0 + col) * K + kc + KC, 0, 0);
#if USE_ASYNC
      __builtin_amdgcn_s_wait_asynccnt(2);      // drain current buffer's loads
#endif
    } else {
#if USE_ASYNC
      __builtin_amdgcn_s_wait_asynccnt(0);
#endif
    }
    __syncthreads();                            // all waves' stage visible

    const char* abuf = asmem + (i & 1) * ABUF + (wm * 16 + col) * ASTRIDE;
    FragU a0, a1;                               // k0 = 0 / 32 within chunk
    a0.q[0] = *(const uint4*)(abuf + khalf * 16);
    a0.q[1] = *(const uint4*)(abuf + khalf * 16 + 32);
    a1.q[0] = *(const uint4*)(abuf + 64 + khalf * 16);
    a1.q[1] = *(const uint4*)(abuf + 64 + khalf * 16 + 32);

    FragU bfr[8];                               // all B frags first: loads overlap
#pragma unroll
    for (int t = 0; t < 4; ++t) {
      const bf16* wp = W + (long)(c0 + t * 16 + col) * K + kc + khalf * 8;
      bfr[t * 2].q[0]     = *(const uint4*)(wp);
      bfr[t * 2].q[1]     = *(const uint4*)(wp + 16);
      bfr[t * 2 + 1].q[0] = *(const uint4*)(wp + 32);
      bfr[t * 2 + 1].q[1] = *(const uint4*)(wp + 48);
    }
#pragma unroll
    for (int t = 0; t < 4; ++t) {
      acc[t] = __builtin_amdgcn_wmma_f32_16x16x32_bf16(false, a0.v, false, bfr[t * 2].v,
                                                       (short)0, acc[t], false, false);
      acc[t] = __builtin_amdgcn_wmma_f32_16x16x32_bf16(false, a1.v, false, bfr[t * 2 + 1].v,
                                                       (short)0, acc[t], false, false);
    }
    __syncthreads();                            // protect buffer before re-stage
  }
#pragma unroll
  for (int t = 0; t < 4; ++t) {
    int c = c0 + t * 16 + col;
    float bv = bias[c];
    float lv = (mode == 2) ? ls[c] : 0.f;
#pragma unroll
    for (int j = 0; j < 8; ++j) {
      int r = r0 + j + 8 * khalf;
      if (r < M) {
        float v = acc[t][j] + bv;
        long idx = (long)r * Nout + c;
        if (mode == 0)      outf[idx] = v;
        else if (mode == 1) outb[idx] = f2bf(v * 0.5f * (1.f + erff(v * 0.70710678118f)));
        else                resid[idx] += v * lv;
      }
    }
  }
}

// ---------------------------------------------------------------- qk-norm + RoPE
__global__ void qk_rope_kernel(const float* __restrict__ qkv,
                               bf16* __restrict__ qo, bf16* __restrict__ ko,
                               bf16* __restrict__ vto,
                               const float* __restrict__ qg, const float* __restrict__ qb,
                               const float* __restrict__ kg, const float* __restrict__ kb,
                               int donr) {
  int n = blockIdx.x, hh = blockIdx.y, b = blockIdx.z;
  int d = threadIdx.x;                       // 0..63
  long hb = (long)(b * HEADS + hh);
  long qidx = hb * (long)NPAD * HD + (long)n * HD + d;
  long vidx = (hb * HD + d) * (long)NPAD + n;
  if (n >= NTOK) { qo[qidx] = f2bf(0.f); ko[qidx] = f2bf(0.f); vto[vidx] = f2bf(0.f); return; }
  long ib = ((long)(b * NTOK + n)) * (3 * DIMC) + hh * HD + d;
  float qv = qkv[ib];
  float kv = qkv[ib + DIMC];
  float vv = qkv[ib + 2 * DIMC];
  __shared__ float red[2][4];
  __shared__ float sqv[HD], skv[HD];
  if (donr) {
    float a0 = qv, a1 = qv * qv, a2 = kv, a3 = kv * kv;
    for (int o = 16; o >= 1; o >>= 1) {
      a0 += __shfl_xor(a0, o); a1 += __shfl_xor(a1, o);
      a2 += __shfl_xor(a2, o); a3 += __shfl_xor(a3, o);
    }
    int wv = d >> 5;
    if ((d & 31) == 0) { red[wv][0]=a0; red[wv][1]=a1; red[wv][2]=a2; red[wv][3]=a3; }
    __syncthreads();
    float qm  = (red[0][0] + red[1][0]) * (1.f / HD);
    float qvr = (red[0][1] + red[1][1]) * (1.f / HD) - qm * qm;
    float km  = (red[0][2] + red[1][2]) * (1.f / HD);
    float kvr = (red[0][3] + red[1][3]) * (1.f / HD) - km * km;
    qv = (qv - qm) * rsqrtf(qvr + LNEPS) * qg[d] + qb[d];
    kv = (kv - km) * rsqrtf(kvr + LNEPS) * kg[d] + kb[d];
    if (n >= 1) {
      sqv[d] = qv; skv[d] = kv;
      __syncthreads();
      int p = n - 1;
      int py = p / GRIDP, px = p % GRIDP;
      int j = d & 15, grp = d >> 4;
      float fr = __powf(10000.f, -(float)j * (1.f / 16.f));
      float ang = ((grp < 2) ? (float)py : (float)px) * fr;
      float sn, cs; __sincosf(ang, &sn, &cs);
      float pq = sqv[d ^ 16], pk = skv[d ^ 16];
      float sgn = (grp & 1) ? sn : -sn;
      qv = qv * cs + sgn * pq;
      kv = kv * cs + sgn * pk;
    }
  }
  qo[qidx]  = f2bf(qv * 0.125f);   // fold HD^-0.5
  ko[qidx]  = f2bf(kv);
  vto[vidx] = f2bf(vv);
}

// ---------------------------------------------------------------- attention
__global__ void attn_kernel(const bf16* __restrict__ q, const bf16* __restrict__ k,
                            const bf16* __restrict__ vt, bf16* __restrict__ o) {
  extern __shared__ __attribute__((aligned(16))) char smem_raw[];
  float* sc = (float*)smem_raw;                    // [16][NPAD] fp32
  bf16*  pb = (bf16*)(smem_raw + 16 * NPAD * 4);   // [16][NPAD] bf16
  int qt = blockIdx.x, hh = blockIdx.y, b = blockIdx.z;
  int tid = threadIdx.x;
  int wave = tid >> 5, lane = tid & 31;
  int col = lane & 15, khalf = lane >> 4;
  long hb = (long)(b * HEADS + hh);
  const bf16* qbase = q  + hb * (long)NPAD * HD;
  const bf16* kbase = k  + hb * (long)NPAD * HD;
  const bf16* vbase = vt + hb * (long)HD * NPAD;

  FragU qa[2];
  {
    const bf16* qp = qbase + (long)(qt * 16 + col) * HD + khalf * 8;
    qa[0].q[0] = *(const uint4*)(qp);
    qa[0].q[1] = *(const uint4*)(qp + 16);
    qa[1].q[0] = *(const uint4*)(qp + 32);
    qa[1].q[1] = *(const uint4*)(qp + 48);
  }
  for (int kt = wave; kt < NPAD / 16; kt += 4) {
    v8f s = {};
    const bf16* kp = kbase + (long)(kt * 16 + col) * HD + khalf * 8;
    FragU kb0, kb1;
    kb0.q[0] = *(const uint4*)(kp);
    kb0.q[1] = *(const uint4*)(kp + 16);
    kb1.q[0] = *(const uint4*)(kp + 32);
    kb1.q[1] = *(const uint4*)(kp + 48);
    s = __builtin_amdgcn_wmma_f32_16x16x32_bf16(false, qa[0].v, false, kb0.v, (short)0, s, false, false);
    s = __builtin_amdgcn_wmma_f32_16x16x32_bf16(false, qa[1].v, false, kb1.v, (short)0, s, false, false);
    int key = kt * 16 + col;
    bool valid = key < NTOK;
#pragma unroll
    for (int j = 0; j < 8; ++j)
      sc[(j + 8 * khalf) * NPAD + key] = valid ? s[j] : -1e30f;
  }
  __syncthreads();
  {
    int row = tid >> 3, sub = tid & 7;
    float m = -1e30f;
    for (int key = sub; key < NPAD; key += 8) m = fmaxf(m, sc[row * NPAD + key]);
    m = fmaxf(m, __shfl_xor(m, 1)); m = fmaxf(m, __shfl_xor(m, 2)); m = fmaxf(m, __shfl_xor(m, 4));
    float sum = 0.f;
    for (int key = sub; key < NPAD; key += 8) {
      float e = __expf(sc[row * NPAD + key] - m);
      sc[row * NPAD + key] = e;
      sum += e;
    }
    sum += __shfl_xor(sum, 1); sum += __shfl_xor(sum, 2); sum += __shfl_xor(sum, 4);
    float inv = 1.f / sum;
    for (int key = sub; key < NPAD; key += 8)
      pb[row * NPAD + key] = f2bf(sc[row * NPAD + key] * inv);
  }
  __syncthreads();
  v8f oacc = {};
  const bf16* vcol = vbase + (long)(wave * 16 + col) * NPAD;
  for (int k0 = 0; k0 < NPAD; k0 += 32) {
    FragU pa, vb;
    const bf16* pp = pb + col * NPAD + k0 + khalf * 8;
    pa.q[0] = *(const uint4*)(pp);
    pa.q[1] = *(const uint4*)(pp + 16);
    const bf16* vp = vcol + k0 + khalf * 8;
    vb.q[0] = *(const uint4*)(vp);
    vb.q[1] = *(const uint4*)(vp + 16);
    oacc = __builtin_amdgcn_wmma_f32_16x16x32_bf16(false, pa.v, false, vb.v, (short)0, oacc, false, false);
  }
#pragma unroll
  for (int j = 0; j < 8; ++j) {
    int r = qt * 16 + j + 8 * khalf;
    if (r < NTOK)
      o[((long)(b * NTOK + r)) * DIMC + hh * HD + wave * 16 + col] = f2bf(oacc[j]);
  }
}

// ---------------------------------------------------------------- host
extern "C" void kernel_launch(void* const* d_in, const int* in_sizes, int n_in,
                              void* d_out, int out_size, void* d_ws, size_t ws_size,
                              hipStream_t stream) {
  const float* x       = (const float*)d_in[0];
  const float* patch_w = (const float*)d_in[1];
  const float* patch_b = (const float*)d_in[2];
  const float* cls_tok = (const float*)d_in[3];
  const float* pos     = (const float*)d_in[4];
  const float* cam     = (const float*)d_in[5];
  const float* n1g     = (const float*)d_in[6];
  const float* n1b     = (const float*)d_in[7];
  const float* qkv_w   = (const float*)d_in[8];
  const float* qkv_b   = (const float*)d_in[9];
  const float* qng     = (const float*)d_in[10];
  const float* qnb     = (const float*)d_in[11];
  const float* kng     = (const float*)d_in[12];
  const float* knb     = (const float*)d_in[13];
  const float* proj_w  = (const float*)d_in[14];
  const float* proj_b  = (const float*)d_in[15];
  const float* ls1     = (const float*)d_in[16];
  const float* n2g     = (const float*)d_in[17];
  const float* n2b     = (const float*)d_in[18];
  const float* fc1_w   = (const float*)d_in[19];
  const float* fc1_b   = (const float*)d_in[20];
  const float* fc2_w   = (const float*)d_in[21];
  const float* fc2_b   = (const float*)d_in[22];
  const float* ls2     = (const float*)d_in[23];
  float* out = (float*)d_out;
  (void)in_sizes; (void)n_in; (void)out_size; (void)ws_size;

  size_t cur = 0;
  char* wsb = (char*)d_ws;
  auto alloc = [&](size_t bytes) -> void* {
    void* p = wsb + cur;
    cur += (bytes + 255) & ~(size_t)255;
    return p;
  };
  const long NQKV = (long)DEPTH * 3 * DIMC * DIMC;
  const long NPRJ = (long)DEPTH * DIMC * DIMC;
  const long NFC  = (long)DEPTH * HID * DIMC;
  bf16*  wq  = (bf16*)alloc(NQKV * 2);
  bf16*  wp  = (bf16*)alloc(NPRJ * 2);
  bf16*  w1  = (bf16*)alloc(NFC * 2);
  bf16*  w2  = (bf16*)alloc(NFC * 2);
  float* h   = (float*)alloc((size_t)MROWS * DIMC * 4);
  bf16*  hn  = (bf16*) alloc((size_t)MROWS * DIMC * 2);
  float* qkvbuf = (float*)alloc((size_t)MROWS * 3 * DIMC * 4);
  bf16*  qb_ = (bf16*)alloc((size_t)BATCH * HEADS * NPAD * HD * 2);
  bf16*  kb_ = (bf16*)alloc((size_t)BATCH * HEADS * NPAD * HD * 2);
  bf16*  vtb = (bf16*)alloc((size_t)BATCH * HEADS * HD * NPAD * 2);
  bf16*  ob  = (bf16*)alloc((size_t)MROWS * DIMC * 2);
  bf16*  mid = (bf16*)alloc((size_t)MROWS * HID * 2);

  (void)hipFuncSetAttribute((const void*)attn_kernel,
                            hipFuncAttributeMaxDynamicSharedMemorySize, ATTN_LDS);

  cvt_kernel<<<2048, 256, 0, stream>>>(qkv_w, wq, NQKV);
  cvt_kernel<<<1024, 256, 0, stream>>>(proj_w, wp, NPRJ);
  cvt_kernel<<<2048, 256, 0, stream>>>(fc1_w, w1, NFC);
  cvt_kernel<<<2048, 256, 0, stream>>>(fc2_w, w2, NFC);

  patch_kernel<<<dim3(NSP, 3, BATCH), 128, 0, stream>>>(x, patch_w, patch_b, pos, h);
  cls_kernel<<<BATCH, DIMC, 0, stream>>>(cls_tok, pos, h);

  int featIdx = 0;
  for (int i = 0; i < DEPTH; ++i) {
    ln_kernel<<<MROWS, 128, 0, stream>>>(h, n1g + i * DIMC, n1b + i * DIMC, hn, cam,
                                         (i == 4) ? 1 : 0);
    gemm_kernel<<<dim3((MROWS + 63) / 64, (3 * DIMC) / 128), 256, 0, stream>>>(
        hn, wq + (long)i * 3 * DIMC * DIMC, MROWS, DIMC, 3 * DIMC,
        qkv_b + (long)i * 3 * DIMC, nullptr, qkvbuf, nullptr, nullptr, 0);
    qk_rope_kernel<<<dim3(NPAD, HEADS, BATCH), HD, 0, stream>>>(
        qkvbuf, qb_, kb_, vtb, qng + i * HD, qnb + i * HD, kng + i * HD, knb + i * HD,
        (i >= 4) ? 1 : 0);
    attn_kernel<<<dim3(NPAD / 16, HEADS, BATCH), 128, ATTN_LDS, stream>>>(qb_, kb_, vtb, ob);
    gemm_kernel<<<dim3((MROWS + 63) / 64, DIMC / 128), 256, 0, stream>>>(
        ob, wp + (long)i * DIMC * DIMC, MROWS, DIMC, DIMC,
        proj_b + (long)i * DIMC, ls1 + (long)i * DIMC, nullptr, nullptr, h, 2);
    ln_kernel<<<MROWS, 128, 0, stream>>>(h, n2g + i * DIMC, n2b + i * DIMC, hn, nullptr, 0);
    gemm_kernel<<<dim3((MROWS + 63) / 64, HID / 128), 256, 0, stream>>>(
        hn, w1 + (long)i * HID * DIMC, MROWS, DIMC, HID,
        fc1_b + (long)i * HID, nullptr, nullptr, mid, nullptr, 1);
    gemm_kernel<<<dim3((MROWS + 63) / 64, DIMC / 128), 256, 0, stream>>>(
        mid, w2 + (long)i * DIMC * HID, MROWS, HID, DIMC,
        fc2_b + (long)i * DIMC, ls2 + (long)i * DIMC, nullptr, nullptr, h, 2);
    if (i == 5 || i == 7 || i == 9 || i == 11) {
      copy_kernel<<<1024, 256, 0, stream>>>(h, out + (long)featIdx * MROWS * DIMC,
                                            (long)MROWS * DIMC);
      ++featIdx;
    }
  }
}
